// CosineAttention_36361193128320
// MI455X (gfx1250) — matrix operand order repaired and checked by reading, not compiled, not run on hardware
//
#include <hip/hip_runtime.h>
#include <hip/hip_bf16.h>

// ---------------------------------------------------------------------------
// CDNA5 (gfx1250) cosine attention, wave32 + v_wmma_f32_16x16x32_bf16
//   + async global->LDS copies (ASYNCcnt) + pos_bias streaming prefetch
// B=2, N=2048, C=1024, H=16, D=64
// ---------------------------------------------------------------------------

#define BATCH 2
#define SEQ   2048
#define CDIM  1024
#define NH    16
#define HD    64
#define QKVN  3072  // 3*NH*HD

typedef __attribute__((ext_vector_type(16))) __bf16 v16bf;
typedef __attribute__((ext_vector_type(8)))  float  v8f;

union Frag16 {
    uint4 u[2];
    v16bf v;
};

__device__ __forceinline__ unsigned short f2bf(float f) {
    unsigned int u = __float_as_uint(f);
    u += 0x7fffu + ((u >> 16) & 1u);   // round-to-nearest-even
    return (unsigned short)(u >> 16);
}

// CDNA5 async DMA: global -> LDS, 16 bytes per lane, tracked by ASYNCcnt.
__device__ __forceinline__ void async_copy_b128(void* lds_dst, const void* gsrc) {
    unsigned lds_addr = (unsigned)(uintptr_t)lds_dst;            // low 32 bits = LDS offset
    unsigned long long gaddr = (unsigned long long)(uintptr_t)gsrc;
    asm volatile("global_load_async_to_lds_b128 %0, %1, off"
                 :: "v"(lds_addr), "v"(gaddr)
                 : "memory");
}
__device__ __forceinline__ void wait_async0() {
    asm volatile("s_wait_asynccnt 0" ::: "memory");
}

// ---------------------------------------------------------------------------
// Kernel 1: fp32 -> bf16 elementwise convert
// ---------------------------------------------------------------------------
__global__ __launch_bounds__(256) void cvt_f32_bf16(const float* __restrict__ in,
                                                    unsigned short* __restrict__ out,
                                                    int n) {
    int i = blockIdx.x * 256 + threadIdx.x;
    if (i < n) out[i] = f2bf(in[i]);
}

// ---------------------------------------------------------------------------
// Kernel 2: bf16 GEMM, C(f32)[M,N] = A[M,K] * B[K,N].  Row-major everywhere.
// 128 threads = 4 waves; block tile 64x64; K step 32; wave computes 16x64.
// A tile moved with async global->LDS b128; B tile transposed via VGPRs.
// ---------------------------------------------------------------------------
__global__ __launch_bounds__(128) void gemm_bf16_wmma(
    const unsigned short* __restrict__ A,
    const unsigned short* __restrict__ Bm,
    float* __restrict__ C,
    int M, int N, int K)
{
    __shared__ alignas(16) unsigned short sA[64 * 40];   // A tile [m][k], stride 40
    __shared__ alignas(16) unsigned short sBt[64 * 40];  // B tile transposed [n][k]

    const int t    = threadIdx.x;
    const int lane = t & 31;
    const int wave = t >> 5;
    const int hh   = lane >> 4;   // half-wave id
    const int ln   = lane & 15;

    const int m0 = blockIdx.y * 64;
    const int n0 = blockIdx.x * 64;

    v8f acc[4] = {};

    for (int k0 = 0; k0 < K; k0 += 32) {
        __syncthreads();
        // A tile 64x32 bf16 = 4 KB: 256 b128 async copies, 2 per thread
        for (int it = 0; it < 2; ++it) {
            int idx = it * 128 + t;
            int r = idx >> 2, c8 = (idx & 3) * 8;
            async_copy_b128(&sA[r * 40 + c8],
                            &A[(size_t)(m0 + r) * K + k0 + c8]);
        }
        // B tile 32x64 -> sBt transposed (VGPR round-trip, 4 bf16 per load)
        for (int it = 0; it < 4; ++it) {
            int idx = it * 128 + t;
            int r = idx >> 4, c4 = (idx & 15) * 4;
            uint2 d = *(const uint2*)&Bm[(size_t)(k0 + r) * N + n0 + c4];
            sBt[(c4 + 0) * 40 + r] = (unsigned short)(d.x & 0xffffu);
            sBt[(c4 + 1) * 40 + r] = (unsigned short)(d.x >> 16);
            sBt[(c4 + 2) * 40 + r] = (unsigned short)(d.y & 0xffffu);
            sBt[(c4 + 3) * 40 + r] = (unsigned short)(d.y >> 16);
        }
        wait_async0();
        __syncthreads();

        // A fragment: rows wave*16 + ln, K halves by hh (ISA 16-bit A layout)
        Frag16 fa;
        const int mrow = wave * 16 + ln;
        fa.u[0] = *(const uint4*)&sA[mrow * 40 + hh * 8];
        fa.u[1] = *(const uint4*)&sA[mrow * 40 + 16 + hh * 8];

        for (int a = 0; a < 4; ++a) {
            Frag16 fb;  // B frag: n = a*16+ln, K halves by hh (same layout as A)
            const int nn = a * 16 + ln;
            fb.u[0] = *(const uint4*)&sBt[nn * 40 + hh * 8];
            fb.u[1] = *(const uint4*)&sBt[nn * 40 + 16 + hh * 8];
            acc[a] = __builtin_amdgcn_wmma_f32_16x16x32_bf16(
                false, fa.v, false, fb.v, (short)0, acc[a], false, false);
        }
    }

    // C/D layout: VGPR r -> M = r + 8*hh ; N = ln
    float* crow = C + (size_t)(m0 + wave * 16 + hh * 8) * N + n0 + ln;
    for (int r = 0; r < 8; ++r) {
        for (int a = 0; a < 4; ++a)
            crow[a * 16] = acc[a][r];
        crow += N;
    }
}

// ---------------------------------------------------------------------------
// Kernel 3: de-interleave qkv (B,N,H,D,3) f32 -> q,k,v bf16 in (B,H,N,D);
// l2-normalize q,k; fold temperature into q.  One wave per (b,n,h) row.
// ---------------------------------------------------------------------------
__global__ __launch_bounds__(256) void qkv_norm_kernel(
    const float* __restrict__ qkv,
    const float* __restrict__ temperature,
    unsigned short* __restrict__ qb,
    unsigned short* __restrict__ kb,
    unsigned short* __restrict__ vb)
{
    const int wg   = blockIdx.x * 8 + (threadIdx.x >> 5);  // global wave id
    const int lane = threadIdx.x & 31;
    const int h = wg & (NH - 1);
    const int n = (wg >> 4) & (SEQ - 1);
    const int b = wg >> 15;

    const float* rowp = qkv + (size_t)(b * SEQ + n) * QKVN + h * (HD * 3);
    const int d0 = lane * 2;

    float q0 = rowp[d0 * 3 + 0],       q1 = rowp[(d0 + 1) * 3 + 0];
    float k0 = rowp[d0 * 3 + 1],       k1 = rowp[(d0 + 1) * 3 + 1];
    float v0 = rowp[d0 * 3 + 2],       v1 = rowp[(d0 + 1) * 3 + 2];

    float sq = q0 * q0 + q1 * q1;
    float sk = k0 * k0 + k1 * k1;
    for (int m = 16; m >= 1; m >>= 1) {
        sq += __shfl_xor(sq, m, 32);
        sk += __shfl_xor(sk, m, 32);
    }
    float qn = fmaxf(sqrtf(sq), 1e-12f);
    float kn = fmaxf(sqrtf(sk), 1e-12f);
    float qs = temperature[0] / qn;   // fold temperature into q
    float ks = 1.0f / kn;

    size_t o = ((size_t)((b * NH + h) * SEQ + n)) * HD + d0;
    *(unsigned int*)&qb[o] = (unsigned int)f2bf(q0 * qs) | ((unsigned int)f2bf(q1 * qs) << 16);
    *(unsigned int*)&kb[o] = (unsigned int)f2bf(k0 * ks) | ((unsigned int)f2bf(k1 * ks) << 16);
    *(unsigned int*)&vb[o] = (unsigned int)f2bf(v0)      | ((unsigned int)f2bf(v1) << 16);
}

// ---------------------------------------------------------------------------
// Kernel 4: flash attention.  One block = (b, h, 64-row tile); 4 waves,
// each wave owns 16 rows.  Online softmax; pos_bias streamed from HBM with
// prefetch of the next tile; K tile via async DMA; V transposed via VGPRs.
// Output (B, N, H*D) bf16.
// ---------------------------------------------------------------------------
__global__ __launch_bounds__(128) void flash_attn_kernel(
    const unsigned short* __restrict__ qb,
    const unsigned short* __restrict__ kb,
    const unsigned short* __restrict__ vb,
    const float* __restrict__ pos_bias,
    const unsigned char* __restrict__ mask,
    unsigned short* __restrict__ ob)
{
    __shared__ alignas(16) unsigned short sK[64 * 72];       // K tile [j][d]
    __shared__ alignas(16) unsigned short sVt[64 * 72];      // V tile transposed [d][j]
    __shared__ alignas(16) unsigned short sP[4 * 16 * 72];   // per-wave P scratch [m][j]

    const int t    = threadIdx.x;
    const int lane = t & 31;
    const int wave = t >> 5;
    const int hh   = lane >> 4;
    const int ln   = lane & 15;

    const int bid   = blockIdx.x;
    const int itile = bid & 31;          // N/64 = 32 row tiles
    const int h     = (bid >> 5) & (NH - 1);
    const int b     = bid >> 9;
    const int i0    = itile * 64;

    const size_t bh = ((size_t)b * NH + h) * SEQ;

    // Q fragments for this wave's 16 rows, kept in registers for the whole loop
    Frag16 fq[2];
    {
        const unsigned short* qrow = qb + (bh + i0 + wave * 16 + ln) * HD;
        for (int c = 0; c < 2; ++c) {
            fq[c].u[0] = *(const uint4*)&qrow[c * 32 + hh * 8];
            fq[c].u[1] = *(const uint4*)&qrow[c * 32 + 16 + hh * 8];
        }
    }

    bool rbad[8];
    for (int r = 0; r < 8; ++r)
        rbad[r] = mask[b * SEQ + i0 + wave * 16 + hh * 8 + r] != 0;

    float mrun[8], lrun[8];
    v8f oacc[4] = {};
    for (int r = 0; r < 8; ++r) { mrun[r] = -3.0e38f; lrun[r] = 0.0f; }

    const float* pbh = pos_bias + (size_t)h * SEQ * SEQ;

    for (int j0 = 0; j0 < SEQ; j0 += 64) {
        __syncthreads();
        // K tile 64x64 bf16 = 8 KB: 512 b128 async copies, 4 per thread
        for (int it = 0; it < 4; ++it) {
            int idx = it * 128 + t;
            int r = idx >> 3, c8 = (idx & 7) * 8;
            async_copy_b128(&sK[r * 72 + c8],
                            &kb[(bh + j0 + r) * HD + c8]);
        }
        // V tile transposed into sVt[d][j] (VGPR round-trip; async cannot transpose)
        for (int it = 0; it < 8; ++it) {
            int idx = it * 128 + t;
            int r = idx >> 4, c4 = (idx & 15) * 4;
            uint2 d = *(const uint2*)&vb[(bh + j0 + r) * HD + c4];
            sVt[(c4 + 0) * 72 + r] = (unsigned short)(d.x & 0xffffu);
            sVt[(c4 + 1) * 72 + r] = (unsigned short)(d.x >> 16);
            sVt[(c4 + 2) * 72 + r] = (unsigned short)(d.y & 0xffffu);
            sVt[(c4 + 3) * 72 + r] = (unsigned short)(d.y >> 16);
        }
        wait_async0();
        __syncthreads();

        // Prefetch next tile's pos_bias stream (global_prefetch_b8)
        if (j0 + 64 < SEQ) {
            int mm = i0 + wave * 16 + ln;                 // one row per lane half-pair
            __builtin_prefetch(&pbh[(size_t)mm * SEQ + j0 + 64 + hh * 32], 0, 1);
        }

        // S = Q K^T : 4 n-subtiles x 2 K-chunks
        v8f sacc[4] = {};
        for (int a = 0; a < 4; ++a) {
            for (int c = 0; c < 2; ++c) {
                Frag16 fk;  // B frag: B[kk][n] = K[a*16+n][c*32+kk] -> K halves by hh (same layout as A)
                fk.u[0] = *(const uint4*)&sK[(a * 16 + ln) * 72 + c * 32 + hh * 8];
                fk.u[1] = *(const uint4*)&sK[(a * 16 + ln) * 72 + c * 32 + 16 + hh * 8];
                sacc[a] = __builtin_amdgcn_wmma_f32_16x16x32_bf16(
                    false, fq[c].v, false, fk.v, (short)0, sacc[a], false, false);
            }
        }

        // bias + mask -> fp32 scores
        bool cbad[4];
        for (int a = 0; a < 4; ++a)
            cbad[a] = mask[b * SEQ + j0 + a * 16 + ln] != 0;

        float p[4][8];
        for (int a = 0; a < 4; ++a) {
            for (int r = 0; r < 8; ++r) {
                int mm = i0 + wave * 16 + hh * 8 + r;
                int nn = j0 + a * 16 + ln;
                float sval = sacc[a][r] + pbh[(size_t)mm * SEQ + nn];
                p[a][r] = (rbad[r] || cbad[a]) ? -3.0e38f : sval;
            }
        }

        // online softmax: butterfly over the 16 lanes sharing each row
        float mnew[8], scale[8];
        for (int r = 0; r < 8; ++r) {
            float mx = fmaxf(fmaxf(p[0][r], p[1][r]), fmaxf(p[2][r], p[3][r]));
            for (int m = 8; m >= 1; m >>= 1)
                mx = fmaxf(mx, __shfl_xor(mx, m, 32));
            mnew[r]  = fmaxf(mrun[r], mx);
            scale[r] = __expf(mrun[r] - mnew[r]);
            mrun[r]  = mnew[r];
        }
        for (int r = 0; r < 8; ++r) {
            float s = 0.0f;
            for (int a = 0; a < 4; ++a) {
                p[a][r] = __expf(p[a][r] - mnew[r]);
                s += p[a][r];
            }
            for (int m = 8; m >= 1; m >>= 1)
                s += __shfl_xor(s, m, 32);
            lrun[r] = lrun[r] * scale[r] + s;
            for (int a = 0; a < 4; ++a)
                oacc[a][r] *= scale[r];
        }

        // P (C-layout) -> per-wave LDS -> A-layout fragments
        unsigned short* pP = &sP[wave * 16 * 72];
        for (int a = 0; a < 4; ++a)
            for (int r = 0; r < 8; ++r)
                pP[(hh * 8 + r) * 72 + a * 16 + ln] = f2bf(p[a][r]);
        asm volatile("s_wait_dscnt 0" ::: "memory");
        __builtin_amdgcn_wave_barrier();

        // O += P V : 2 j-chunks x 4 d-subtiles
        for (int c = 0; c < 2; ++c) {
            Frag16 fp;
            fp.u[0] = *(const uint4*)&pP[ln * 72 + c * 32 + hh * 8];
            fp.u[1] = *(const uint4*)&pP[ln * 72 + c * 32 + 16 + hh * 8];
            for (int a = 0; a < 4; ++a) {
                Frag16 fv;  // B[kk][n] = V[c*32+kk][a*16+n] = sVt[a*16+n][c*32+kk]
                fv.u[0] = *(const uint4*)&sVt[(a * 16 + ln) * 72 + c * 32 + hh * 8];
                fv.u[1] = *(const uint4*)&sVt[(a * 16 + ln) * 72 + c * 32 + 16 + hh * 8];
                oacc[a] = __builtin_amdgcn_wmma_f32_16x16x32_bf16(
                    false, fp.v, false, fv.v, (short)0, oacc[a], false, false);
            }
        }
    }

    // epilogue: O / l, write (B, N, H*D) bf16
    for (int r = 0; r < 8; ++r) {
        float inv = 1.0f / lrun[r];
        int mm = i0 + wave * 16 + hh * 8 + r;
        for (int a = 0; a < 4; ++a) {
            size_t off = ((size_t)(b * SEQ + mm)) * (NH * HD) + h * HD + a * 16 + ln;
            ob[off] = f2bf(oacc[a][r] * inv);
        }
    }
}

// ---------------------------------------------------------------------------
// launch
// ---------------------------------------------------------------------------
extern "C" void kernel_launch(void* const* d_in, const int* in_sizes, int n_in,
                              void* d_out, int out_size, void* d_ws, size_t ws_size,
                              hipStream_t stream) {
    const float*         x    = (const float*)d_in[0];          // (B,N,C)
    const float*         Wqkv = (const float*)d_in[1];          // (C, 3HD)
    const float*         Wout = (const float*)d_in[2];          // (HD, C)
    const float*         temp = (const float*)d_in[3];          // scalar
    const float*         pb   = (const float*)d_in[4];          // (H,N,N)
    const unsigned char* mask = (const unsigned char*)d_in[5];  // (B,N) bool

    char* ws = (char*)d_ws;
    const size_t NX    = (size_t)BATCH * SEQ * CDIM;      // 4 Mi elems
    const size_t NWQKV = (size_t)CDIM * QKVN;             // 3 Mi elems
    const size_t NWOUT = (size_t)CDIM * CDIM;             // 1 Mi elems
    const size_t NQKV  = (size_t)BATCH * SEQ * QKVN;      // 12 Mi elems
    const size_t NHEAD = (size_t)BATCH * NH * SEQ * HD;   // 4 Mi elems

    unsigned short* xbf    = (unsigned short*)(ws);
    unsigned short* wqkvbf = (unsigned short*)(ws + NX * 2);
    unsigned short* woutbf = (unsigned short*)(ws + NX * 2 + NWQKV * 2);
    float*          qkvf   = (float*)(ws + NX * 2 + NWQKV * 2 + NWOUT * 2);
    char*           p2     = ws + NX * 2 + NWQKV * 2 + NWOUT * 2 + NQKV * 4;
    unsigned short* qb  = (unsigned short*)(p2);
    unsigned short* kb  = (unsigned short*)(p2 + NHEAD * 2);
    unsigned short* vb  = (unsigned short*)(p2 + NHEAD * 4);
    unsigned short* obf = (unsigned short*)(p2 + NHEAD * 6);

    // 1) convert to bf16
    cvt_f32_bf16<<<(int)((NX    + 255) / 256), 256, 0, stream>>>(x,    xbf,    (int)NX);
    cvt_f32_bf16<<<(int)((NWQKV + 255) / 256), 256, 0, stream>>>(Wqkv, wqkvbf, (int)NWQKV);
    cvt_f32_bf16<<<(int)((NWOUT + 255) / 256), 256, 0, stream>>>(Wout, woutbf, (int)NWOUT);

    // 2) QKV projection: (4096 x 1024) @ (1024 x 3072) -> f32
    gemm_bf16_wmma<<<dim3(QKVN / 64, (BATCH * SEQ) / 64), 128, 0, stream>>>(
        xbf, wqkvbf, qkvf, BATCH * SEQ, QKVN, CDIM);

    // 3) split + l2norm + temperature fold, to (B,H,N,D) bf16
    qkv_norm_kernel<<<(BATCH * SEQ * NH) / 8, 256, 0, stream>>>(qkvf, temp, qb, kb, vb);

    // 4) flash attention -> (B, N, H*D) bf16
    flash_attn_kernel<<<BATCH * NH * (SEQ / 64), 128, 0, stream>>>(qb, kb, vb, pb, mask, obf);

    // 5) output projection: (4096 x 1024) @ (1024 x 1024) -> d_out f32
    gemm_bf16_wmma<<<dim3(CDIM / 64, (BATCH * SEQ) / 64), 128, 0, stream>>>(
        obf, woutbf, (float*)d_out, BATCH * SEQ, CDIM, CDIM);
}
